// MoCEAdapter_28235115003999
// MI455X (gfx1250) — matrix operand order repaired and checked by reading, not compiled, not run on hardware
//
#include <hip/hip_runtime.h>
#include <hip/hip_bf16.h>
#include <math.h>

// ---------------- WMMA types (CDNA5, wave32) ----------------
typedef __bf16 bf16_t;
typedef bf16_t v16bf __attribute__((ext_vector_type(16)));
typedef float  v8f   __attribute__((ext_vector_type(8)));

// =====================================================================
// conv1x1 as GEMM via v_wmma_f32_16x16x32_bf16.
// X: [B, K, HW] (NCHW), W: [O, K], Y: [B, O, HW]
// Each wave owns one 16-position n-tile, loads the A fragments ONCE
// (K <= 64 -> at most 2 v16bf fragments) and loops over all o-tiles,
// reusing A. Weights are tiny (L0/L2 resident) and loaded as float4.
// Epilogue: bias, SiLU (fast v_rcp), residual, per-batch gate, accumulate.
// =====================================================================
__global__ void __launch_bounds__(128)
conv1x1_wmma(const float* __restrict__ X, const float* __restrict__ Wt,
             const float* __restrict__ bias, const float* __restrict__ resid,
             const float* __restrict__ gates, int gate_idx,
             float* __restrict__ Y,
             int K, int O, int HW, int epi, int accum,
             long long total_ntiles, int tiles_o)
{
    int wave = threadIdx.x >> 5;
    int lane = threadIdx.x & 31;
    long long t = (long long)blockIdx.x * 4 + wave;
    if (t >= total_ntiles) return;            // wave-uniform: EXEC stays all-1s
    long long n0 = t * 16;                    // HW % 16 == 0 -> never crosses batch
    int b   = (int)(n0 / HW);
    int hw0 = (int)(n0 % HW);
    const float* Xb = X + ((size_t)b * K) * HW + hw0;

    int m  = lane & 15;     // A: M row / B: N col
    int hi = lane >> 4;
    int nk = K >> 5;        // 1 or 2 k-blocks (uniform)

    // ---- load A fragments once, reuse across all o-tiles ----
    // A 16x32 bf16 layout: lanes 0-15 -> M=lane, K=base..base+15;
    // lanes 16-31 -> K=base+16..base+31 (matches B).
    v16bf afrag[2];
#pragma unroll
    for (int kk = 0; kk < 2; ++kk) {
        if (kk < nk) {
            int kA = (kk << 5) + (hi << 4);
#pragma unroll
            for (int i = 0; i < 16; ++i) {
                int k = kA + i;
                afrag[kk][i] = (bf16_t)Xb[(size_t)k * HW + m];
            }
        }
    }

    float g = gates ? gates[b * 4 + gate_idx] : 1.0f;

    for (int ot = 0; ot < tiles_o; ++ot) {
        int oc = ot * 16 + m;
        v8f acc = {};
#pragma unroll
        for (int kk = 0; kk < 2; ++kk) {
            if (kk >= nk) break;              // uniform
            // B 32x16 bf16: lanes 0-15: N=lane, K=0..15; lanes 16-31: K=16..31
            v16bf bv;
            const float4* wr4 = (const float4*)(Wt + (size_t)oc * K + (kk << 5) + (hi << 4));
#pragma unroll
            for (int i = 0; i < 4; ++i) {
                float4 w4 = wr4[i];
                bv[4 * i + 0] = (bf16_t)w4.x;
                bv[4 * i + 1] = (bf16_t)w4.y;
                bv[4 * i + 2] = (bf16_t)w4.z;
                bv[4 * i + 3] = (bf16_t)w4.w;
            }
            acc = __builtin_amdgcn_wmma_f32_16x16x32_bf16(
                false, afrag[kk], false, bv, (short)0, acc, false, false);
        }
        float bias_v = bias ? bias[oc] : 0.0f;
        size_t ybase = ((size_t)b * O + oc) * HW + hw0;
#pragma unroll
        for (int r = 0; r < 8; ++r) {
            int mm = r + (hi << 3);
            float v = acc[r] + bias_v;
            if (epi == 1) v = v * __builtin_amdgcn_rcpf(1.0f + __expf(-v)); // fast SiLU
            if (resid) v += resid[ybase + mm];
            v *= g;
            if (accum) Y[ybase + mm] += v;
            else       Y[ybase + mm]  = v;
        }
    }
}

// ---------------- depthwise 3x3, same padding ----------------
__global__ void dwconv3x3(const float* __restrict__ X, const float* __restrict__ Wd,
                          const float* __restrict__ bias, float* __restrict__ Y,
                          int B, int C, int H, int W)
{
    size_t idx = (size_t)blockIdx.x * blockDim.x + threadIdx.x;
    size_t total = (size_t)B * C * H * W;
    if (idx >= total) return;
    int wpos = (int)(idx % W); size_t t = idx / W;
    int hpos = (int)(t % H);   t /= H;
    int c = (int)(t % C); int b = (int)(t / C);
    const float* xb = X + ((size_t)b * C + c) * H * W;
    const float* wk = Wd + c * 9;
    float s = bias ? bias[c] : 0.0f;
#pragma unroll
    for (int di = -1; di <= 1; ++di) {
        int hh = hpos + di;
        if (hh < 0 || hh >= H) continue;
#pragma unroll
        for (int dj = -1; dj <= 1; ++dj) {
            int ww = wpos + dj;
            if (ww < 0 || ww >= W) continue;
            s += xb[(size_t)hh * W + ww] * wk[(di + 1) * 3 + (dj + 1)];
        }
    }
    Y[idx] = s;
}

// ---------------- shared attention: Gram matrix + row norms ----------------
// block = b*8+h (256 thr). Produces per (b,h): qq[8], kk[8], S[8][8].
__global__ void __launch_bounds__(256)
attn_gram(const float* __restrict__ QKV, float* __restrict__ Sout, int HW)
{
    int bh = blockIdx.x;
    int b = bh >> 3, h = bh & 7;
    const float* q = QKV + ((size_t)b * 192 + h * 8) * HW;
    const float* k = QKV + ((size_t)b * 192 + 64 + h * 8) * HW;
    int c = threadIdx.x & 7;
    int g = threadIdx.x >> 3;   // 0..31
    float s[8] = {}, qq = 0.f, kk = 0.f;
    for (int n = g; n < HW; n += 32) {
        float qc = q[(size_t)c * HW + n];
        float kc = k[(size_t)c * HW + n];
        qq += qc * qc; kk += kc * kc;
#pragma unroll
        for (int d = 0; d < 8; ++d) s[d] += qc * k[(size_t)d * HW + n];
    }
    __shared__ float lds[8][32][10];
#pragma unroll
    for (int d = 0; d < 8; ++d) lds[c][g][d] = s[d];
    lds[c][g][8] = qq; lds[c][g][9] = kk;
    __syncthreads();
    int tid = threadIdx.x;
    if (tid < 80) {
        int cc = tid / 10, j = tid % 10;
        float acc = 0.f;
        for (int gg = 0; gg < 32; ++gg) acc += lds[cc][gg][j];
        float* o = Sout + (size_t)bh * 80;
        if (j < 8)       o[16 + cc * 8 + j] = acc;  // S[c][d]
        else if (j == 8) o[cc]              = acc;  // ||q_c||^2
        else             o[8 + cc]          = acc;  // ||k_c||^2
    }
}

// normalize + temp + row softmax (8x8 per head)
__global__ void attn_softmax(const float* __restrict__ Sin, const float* __restrict__ temp,
                             float* __restrict__ attn)
{
    int bh = blockIdx.x;
    int h = bh & 7;
    int tid = threadIdx.x;          // 64
    int c = tid >> 3, d = tid & 7;
    const float* S = Sin + (size_t)bh * 80;
    float qn = sqrtf(S[c]);     qn = fmaxf(qn, 1e-12f);
    float kn = sqrtf(S[8 + d]); kn = fmaxf(kn, 1e-12f);
    float a = S[16 + c * 8 + d] / (qn * kn) * temp[h];
    __shared__ float row[8][8];
    row[c][d] = a;
    __syncthreads();
    float mx = row[c][0];
#pragma unroll
    for (int i = 1; i < 8; ++i) mx = fmaxf(mx, row[c][i]);
    float e = __expf(a - mx);
    __shared__ float er[8][8];
    er[c][d] = e;
    __syncthreads();
    float sum = 0.f;
#pragma unroll
    for (int i = 0; i < 8; ++i) sum += er[c][i];
    attn[(size_t)bh * 64 + c * 8 + d] = e / sum;
}

// out[b, h*8+c, n] = sum_d attn[b,h,c,d] * v[b, 128+h*8+d, n]
__global__ void attn_apply(const float* __restrict__ QKV, const float* __restrict__ attn,
                           float* __restrict__ Y, int HW)
{
    size_t idx = (size_t)blockIdx.x * blockDim.x + threadIdx.x;
    size_t total = (size_t)8 * 64 * HW;
    if (idx >= total) return;
    int n = (int)(idx % HW); size_t t = idx / HW;
    int cg = (int)(t % 64); int b = (int)(t / 64);
    int h = cg >> 3, c = cg & 7;
    const float* v = QKV + ((size_t)b * 192 + 128 + h * 8) * HW;
    const float* ar = attn + ((size_t)(b * 8 + h)) * 64 + c * 8;
    float s = 0.f;
#pragma unroll
    for (int d = 0; d < 8; ++d) s += ar[d] * v[(size_t)d * HW + n];
    Y[idx] = s;
}

// ---------------- irfft2(rfft2(q)*rfft2(k)) == 2D 8x8 circular convolution ---------
__global__ void __launch_bounds__(64)
circconv8(const float* __restrict__ Q, const float* __restrict__ Kt, float* __restrict__ O,
          int Cq, int Ck, int H, int W)
{
    int pw = blockIdx.x % (W / 8); int t = blockIdx.x / (W / 8);
    int ph = t % (H / 8); t /= (H / 8);
    int c = t % Cq; int b = t / Cq;
    size_t qbase = ((size_t)b * Cq + c) * H * W + (size_t)ph * 8 * W + (size_t)pw * 8;
    size_t kbase = ((size_t)b * Ck + c) * H * W + (size_t)ph * 8 * W + (size_t)pw * 8;
    __shared__ float qs[64], ks[64];
    int tid = threadIdx.x;
    int i = tid >> 3, j = tid & 7;
    qs[tid] = Q[qbase + (size_t)i * W + j];
    ks[tid] = Kt[kbase + (size_t)i * W + j];
    __syncthreads();
    float s = 0.f;
#pragma unroll
    for (int u = 0; u < 8; ++u)
#pragma unroll
        for (int v = 0; v < 8; ++v)
            s += qs[u * 8 + v] * ks[((i - u) & 7) * 8 + ((j - v) & 7)];
    O[qbase + (size_t)i * W + j] = s;
}

// ---------------- LayerNorm over channel dim, in place ----------------
__global__ void ln_channels_k(float* __restrict__ X, const float* __restrict__ w,
                              const float* __restrict__ bb, int B, int C, int HW)
{
    size_t idx = (size_t)blockIdx.x * blockDim.x + threadIdx.x;
    if (idx >= (size_t)B * HW) return;
    int n = (int)(idx % HW); int b = (int)(idx / HW);
    float* base = X + (size_t)b * C * HW + n;
    float mu = 0.f;
    for (int c = 0; c < C; ++c) mu += base[(size_t)c * HW];
    mu /= (float)C;
    float var = 0.f;
    for (int c = 0; c < C; ++c) { float d = base[(size_t)c * HW] - mu; var += d * d; }
    var /= (float)C;
    float inv = rsqrtf(var + 1e-5f);
    for (int c = 0; c < C; ++c)
        base[(size_t)c * HW] = (base[(size_t)c * HW] - mu) * inv * w[c] + bb[c];
}

// Y[b,c,n] *= V[b, coff+c, n] with channel-count Cv for V
__global__ void mul_chan(float* __restrict__ Y, const float* __restrict__ V,
                         int Cy, int Cv, int coff, int HW)
{
    size_t idx = (size_t)blockIdx.x * blockDim.x + threadIdx.x;
    size_t total = (size_t)8 * Cy * HW;
    if (idx >= total) return;
    int n = (int)(idx % HW); size_t t = idx / HW;
    int c = (int)(t % Cy); int b = (int)(t / Cy);
    Y[idx] *= V[((size_t)b * Cv + coff + c) * HW + n];
}

// ---------------- freq embedding: fixed highpass 3x3 -> gelu -> spatial mean ----
__global__ void __launch_bounds__(256)
hp_gelu_mean(const float* __restrict__ X, float* __restrict__ out, int C, int H, int W)
{
    int bc = blockIdx.x; int c = bc % C; int b = bc / C;
    const float* xb = X + ((size_t)b * C + c) * H * W;
    float acc = 0.f;
    for (int p = threadIdx.x; p < H * W; p += blockDim.x) {
        int i = p / W, j = p % W;
        float s = 8.f * xb[p];
        for (int di = -1; di <= 1; ++di)
            for (int dj = -1; dj <= 1; ++dj) {
                if (di == 0 && dj == 0) continue;
                int hh = i + di, ww = j + dj;
                if (hh < 0 || hh >= H || ww < 0 || ww >= W) continue;
                s -= xb[(size_t)hh * W + ww];
            }
        acc += 0.5f * s * (1.f + erff(s * 0.70710678118654752f));  // exact gelu
    }
    __shared__ float red[256];
    red[threadIdx.x] = acc;
    __syncthreads();
    for (int st = 128; st > 0; st >>= 1) {
        if (threadIdx.x < st) red[threadIdx.x] += red[threadIdx.x + st];
        __syncthreads();
    }
    if (threadIdx.x == 0) out[bc] = red[0] / (float)(H * W);
}

__global__ void __launch_bounds__(256)
chan_mean(const float* __restrict__ X, float* __restrict__ out, int C, int H, int W)
{
    int bc = blockIdx.x; int c = bc % C; int b = bc / C;
    const float* xb = X + ((size_t)b * C + c) * H * W;
    float acc = 0.f;
    for (int p = threadIdx.x; p < H * W; p += blockDim.x) acc += xb[p];
    __shared__ float red[256];
    red[threadIdx.x] = acc;
    __syncthreads();
    for (int st = 128; st > 0; st >>= 1) {
        if (threadIdx.x < st) red[threadIdx.x] += red[threadIdx.x + st];
        __syncthreads();
    }
    if (threadIdx.x == 0) out[bc] = red[0] / (float)(H * W);
}

// ---------------- freq MLP + routing softmax + top-2 (tiny, one block) --------
__global__ void __launch_bounds__(256)
routing_k(const float* __restrict__ pooled, const float* __restrict__ hpm,
          const float* __restrict__ w1, const float* __restrict__ b1,
          const float* __restrict__ w2, const float* __restrict__ b2,
          const float* __restrict__ gate_w, const float* __restrict__ fg_w,
          const float* __restrict__ noise, float* __restrict__ gates)
{
    __shared__ float hid[8 * 128];
    __shared__ float fe[8 * 64];
    int tid = threadIdx.x;
    for (int u = tid; u < 8 * 128; u += blockDim.x) {
        int b = u / 128, o = u % 128;
        float s = b1[o];
        for (int c = 0; c < 64; ++c) s += hpm[b * 64 + c] * w1[o * 64 + c];
        hid[u] = 0.5f * s * (1.f + erff(s * 0.70710678118654752f));
    }
    __syncthreads();
    for (int u = tid; u < 8 * 64; u += blockDim.x) {
        int b = u / 64, o = u % 64;
        float s = b2[o];
        for (int c = 0; c < 128; ++c) s += hid[b * 128 + c] * w2[o * 128 + c];
        fe[u] = s;
    }
    __syncthreads();
    if (tid < 8) {
        int b = tid;
        float lg[4];
        for (int e = 0; e < 4; ++e) {
            float s = 0.f;
            for (int c = 0; c < 64; ++c)
                s += pooled[b * 64 + c] * gate_w[e * 64 + c]
                   + fe[b * 64 + c]     * fg_w[e * 64 + c];
            lg[e] = s + noise[b * 4 + e] * 0.25f;   // NOISE_STD = 1/NUM_EXPERTS
        }
        float mx = fmaxf(fmaxf(lg[0], lg[1]), fmaxf(lg[2], lg[3]));
        float ex[4], sum = 0.f;
        for (int e = 0; e < 4; ++e) { ex[e] = __expf(lg[e] - mx); sum += ex[e]; }
        float sc[4];
        for (int e = 0; e < 4; ++e) sc[e] = ex[e] / sum;
        int i0 = 0;
        for (int e = 1; e < 4; ++e) if (sc[e] > sc[i0]) i0 = e;
        int i1 = (i0 == 0) ? 1 : 0;
        for (int e = 0; e < 4; ++e) if (e != i0 && sc[e] > sc[i1]) i1 = e;
        for (int e = 0; e < 4; ++e)
            gates[b * 4 + e] = (e == i0 || e == i1) ? sc[e] : 0.f;
    }
}

// =====================================================================
extern "C" void kernel_launch(void* const* d_in, const int* in_sizes, int n_in,
                              void* d_out, int out_size, void* d_ws, size_t ws_size,
                              hipStream_t stream)
{
    if (n_in < 26) return;
    const int Bc = 8, C = 64, Hh = 192, Ww = 192, HW = Hh * Ww, Rr = 32, E = 4;
    const size_t N = (size_t)Bc * HW;   // 294912 spatial positions

    const float* x       = (const float*)d_in[0];
    const float* noise   = (const float*)d_in[1];
    const float* qkv_w   = (const float*)d_in[2];
    const float* qkv_b   = (const float*)d_in[3];
    const float* qkv_dww = (const float*)d_in[4];
    const float* qkv_dwb = (const float*)d_in[5];
    const float* po_w    = (const float*)d_in[6];
    const float* po_b    = (const float*)d_in[7];
    const float* temp    = (const float*)d_in[8];
    const float* f_w1    = (const float*)d_in[9];
    const float* f_b1    = (const float*)d_in[10];
    const float* f_w2    = (const float*)d_in[11];
    const float* f_b2    = (const float*)d_in[12];
    const float* gate_w  = (const float*)d_in[13];
    const float* fg_w    = (const float*)d_in[14];
    const float* e_p0    = (const float*)d_in[15];
    const float* e_p1    = (const float*)d_in[16];
    const float* e_p2    = (const float*)d_in[17];
    const float* e_qw    = (const float*)d_in[18];
    const float* e_qdw   = (const float*)d_in[19];
    const float* e_kvw   = (const float*)d_in[20];
    const float* e_kvdw  = (const float*)d_in[21];
    const float* e_lnw   = (const float*)d_in[22];
    const float* e_lnb   = (const float*)d_in[23];
    const float* e_pow   = (const float*)d_in[24];
    const float* e_pob   = (const float*)d_in[25];
    float* out = (float*)d_out;

    // --- workspace (bump allocation with lifetime-based aliasing) ---
    float* W1   = (float*)d_ws;                      // 8*192*HW floats
    float* W2   = W1 + (size_t)Bc * 192 * HW;        // 8*192*HW floats
    float* tail = W2 + (size_t)Bc * 192 * HW;

    float* qkv_raw  = W1;
    float* qkv_dw   = W2;
    float* attn_out = W1;                 // qkv_raw dead after dwconv
    float* shared   = W1 + N * C;
    float* spare    = W1 + 2 * N * C;     // kv_dw lives here
    float* fbuf     = W2;                 // W2 free once `shared` is computed
    float* q_raw    = W2 + N * Rr;
    float* q_dw     = W2 + 2 * N * Rr;
    float* kv_raw   = W2 + 3 * N * Rr;    // 2*N*Rr
    float* kv_dw    = spare;              // 2*N*Rr
    float* obuf     = q_raw;              // q_raw dead after q dwconv
    float* s1       = kv_raw;             // kv_raw dead after kv dwconv
    float* po_out   = kv_raw + N * Rr;

    float* Sg     = tail;                 // 64*80
    float* attnW  = Sg + 64 * 80;         // 64*64
    float* hpm    = attnW + 64 * 64;      // 512
    float* pooled = hpm + 512;            // 512
    float* gates  = pooled + 512;         // 32

    auto gemm = [&](const float* X, const float* Wm, const float* bias,
                    const float* resid, const float* gts, int gidx,
                    float* Y, int K, int O, int epi, int accum) {
        long long ntiles = (long long)(N / 16);   // one wave per 16-position tile
        int tiles_o = O / 16;
        int blocks = (int)((ntiles + 3) / 4);
        conv1x1_wmma<<<blocks, 128, 0, stream>>>(X, Wm, bias, resid, gts, gidx,
                                                 Y, K, O, HW, epi, accum,
                                                 ntiles, tiles_o);
    };

    // ---------- shared attention ----------
    gemm(x, qkv_w, qkv_b, nullptr, nullptr, 0, qkv_raw, C, 3 * C, 0, 0);
    {
        size_t tot = (size_t)Bc * 192 * HW;
        dwconv3x3<<<(unsigned)((tot + 255) / 256), 256, 0, stream>>>(
            qkv_raw, qkv_dww, qkv_dwb, qkv_dw, Bc, 192, Hh, Ww);
    }
    attn_gram<<<64, 256, 0, stream>>>(qkv_dw, Sg, HW);
    attn_softmax<<<64, 64, 0, stream>>>(Sg, temp, attnW);
    {
        size_t tot = N * C;
        attn_apply<<<(unsigned)((tot + 255) / 256), 256, 0, stream>>>(qkv_dw, attnW, attn_out, HW);
    }
    gemm(attn_out, po_w, po_b, nullptr, nullptr, 0, shared, C, C, 0, 0);

    // ---------- routing inputs ----------
    hp_gelu_mean<<<Bc * C, 256, 0, stream>>>(x, hpm, C, Hh, Ww);
    chan_mean<<<Bc * C, 256, 0, stream>>>(x, pooled, C, Hh, Ww);
    routing_k<<<1, 256, 0, stream>>>(pooled, hpm, f_w1, f_b1, f_w2, f_b2,
                                     gate_w, fg_w, noise, gates);

    // ---------- experts (dense eval, gate-weighted combine into d_out) ----------
    for (int e = 0; e < E; ++e) {
        gemm(x, e_p0 + (size_t)e * Rr * C, nullptr, nullptr, nullptr, 0, fbuf, C, Rr, 0, 0);
        gemm(fbuf, e_qw + (size_t)e * Rr * Rr, nullptr, nullptr, nullptr, 0, q_raw, Rr, Rr, 0, 0);
        {
            size_t tot = N * Rr;
            dwconv3x3<<<(unsigned)((tot + 255) / 256), 256, 0, stream>>>(
                q_raw, e_qdw + (size_t)e * Rr * 9, nullptr, q_dw, Bc, Rr, Hh, Ww);
        }
        gemm(fbuf, e_kvw + (size_t)e * 2 * Rr * Rr, nullptr, nullptr, nullptr, 0, kv_raw, Rr, 2 * Rr, 0, 0);
        {
            size_t tot = N * 2 * Rr;
            dwconv3x3<<<(unsigned)((tot + 255) / 256), 256, 0, stream>>>(
                kv_raw, e_kvdw + (size_t)e * 2 * Rr * 9, nullptr, kv_dw, Bc, 2 * Rr, Hh, Ww);
        }
        {
            int nb = Bc * Rr * (Hh / 8) * (Ww / 8);
            circconv8<<<nb, 64, 0, stream>>>(q_dw, kv_dw, obuf, Rr, 2 * Rr, Hh, Ww);
        }
        ln_channels_k<<<(unsigned)((N + 255) / 256), 256, 0, stream>>>(
            obuf, e_lnw + e * Rr, e_lnb + e * Rr, Bc, Rr, HW);
        {
            size_t tot = N * Rr;   // o *= v   (v = second half of kv channels)
            mul_chan<<<(unsigned)((tot + 255) / 256), 256, 0, stream>>>(
                obuf, kv_dw, Rr, 2 * Rr, Rr, HW);
        }
        gemm(obuf, e_pow + (size_t)e * Rr * Rr, e_pob + e * Rr, nullptr, nullptr, 0, po_out, Rr, Rr, 0, 0);
        gemm(shared, e_p1 + (size_t)e * Rr * C, nullptr, nullptr, nullptr, 0, s1, C, Rr, /*silu*/1, 0);
        {
            size_t tot = N * Rr;   // f = att_out * silu(gate)
            mul_chan<<<(unsigned)((tot + 255) / 256), 256, 0, stream>>>(
                po_out, s1, Rr, Rr, 0, HW);
        }
        // out (e==0: =, else: +=)  g[b,e] * (p2(f) + x)
        gemm(po_out, e_p2 + (size_t)e * C * Rr, nullptr, x, gates, e, out, Rr, C, 0, e > 0 ? 1 : 0);
    }
    (void)in_sizes; (void)out_size; (void)ws_size;
}
